// BenchmarkCell_3169685865038
// MI455X (gfx1250) — matrix operand order repaired and checked by reading, not compiled, not run on hardware
//
#include <hip/hip_runtime.h>
#include <hip/hip_bf16.h>
#include <stdint.h>

// ---------------- vector types ----------------
typedef __attribute__((ext_vector_type(16))) __bf16 v16bf;
typedef __attribute__((ext_vector_type(8)))  __bf16 v8bf;
typedef __attribute__((ext_vector_type(8)))  float  v8f;
typedef __attribute__((ext_vector_type(4)))  unsigned int u32x4;
typedef __attribute__((ext_vector_type(8)))  int          i32x8;
typedef __attribute__((ext_vector_type(4)))  int          i32x4;

// ---------------- problem geometry ----------------
#define T_STEPS  70
#define BATCH    256
#define NHID     850
#define NPAD     864            // per-half padded columns (54 tiles of 16)
#define NT_HALF  54
#define NT_FULL  108
#define KPAD1    864            // edge K padded (27 k-tiles of 32)
#define KT1      27
#define KPAD0    1728           // stage0 K padded (x:864 | h:864)
#define KT0      54
#define BH       (BATCH * NHID) // 217600

#define NBLOCKS  108            // 54 np-columns x 2 mt-groups
#define NTHREADS 256            // 8 waves; wave w -> mt = mtg*8 + w

#define W0R_ELEMS  ((size_t)KT0 * NT_FULL * 512)   // bf16 elems
#define WSR_PER    ((size_t)KT1 * NT_FULL * 512)
#define WSR_ELEMS  (8 * WSR_PER)

#define FRAG_USH   512                     // 1 fragment = 32 lanes x 16 bf16 = 1KB
#define ROWSTRIDE_USH ((size_t)NT_FULL * FRAG_USH)  // 55296 ushorts (110592B) between k-tiles
#define EDGE_SLAB_USH (KT1 * FRAG_USH)     // 13824 ushorts = one half-slab (27KB)
#define EDGE_BUF_USH  (2 * EDGE_SLAB_USH)  // 27648 ushorts = c+h (55KB)
#define LDS_BYTES  (2 * EDGE_BUF_USH * 2)  // 110592 bytes (also = stage0 slab)

// ---------------- helpers ----------------
__device__ __forceinline__ unsigned short f2bf(float f) {
  unsigned u = __float_as_uint(f);
  u += 0x7FFFu + ((u >> 16) & 1u);      // round-to-nearest-even
  return (unsigned short)(u >> 16);
}
__device__ __forceinline__ float sigf(float x) { return 1.0f / (1.0f + __expf(-x)); }
__device__ __forceinline__ float actf(int code, float x) {
  switch (code) {
    case 0: return tanhf(x);
    case 1: return fmaxf(x, 0.0f);
    case 2: return sigf(x);
    default: return x;
  }
}

// device-wide barrier for the persistent kernel (108 WGs co-resident, <=1/WGP)
__device__ __forceinline__ void grid_sync(unsigned* cnt, volatile unsigned* gen, unsigned& mygen) {
  __syncthreads();
  if (threadIdx.x == 0) {
    __threadfence();
    if (atomicAdd(cnt, 1u) == (unsigned)(NBLOCKS - 1)) {
      atomicExch(cnt, 0u);
      __threadfence();
      atomicAdd((unsigned*)gen, 1u);
    } else {
      while (*gen <= mygen) { __builtin_amdgcn_s_sleep(2); }
    }
  }
  ++mygen;
  __syncthreads();
}

// ---------------- TDM: async load of one B half-slab into LDS ----------------
// slab = KT rows; row kt is the 1KB fragment at gsrc + kt*ROWSTRIDE, packed
// contiguously into LDS at ldsUshortOff.
#if defined(__has_builtin)
#if __has_builtin(__builtin_amdgcn_tensor_load_to_lds)
#define USE_TDM 1
#endif
#endif

__device__ __forceinline__ void tdm_load_slab(const unsigned short* gsrc,
                                              unsigned ldsUshortOff, int KT) {
  extern __shared__ unsigned short smem[];
#ifdef USE_TDM
  unsigned long long ga = (unsigned long long)(uintptr_t)gsrc;
  unsigned ldsByte = ((unsigned)(uintptr_t)smem) + ldsUshortOff * 2u;
  u32x4 g0;
  g0[0] = 1u;                                            // count=1, user mode
  g0[1] = ldsByte;                                       // lds_addr (bytes)
  g0[2] = (unsigned)(ga & 0xFFFFFFFFu);                  // global_addr[31:0]
  g0[3] = (unsigned)((ga >> 32) & 0x1FFFFFFu) | (2u << 30); // addr[56:32] | type=2
  i32x8 g1;
  g1[0] = 0x30000;                 // workgroup_mask=0, data_size=3 (8B units)
  g1[1] = (int)(128u << 16);       // tensor_dim0[15:0] in bits[63:48]  (128 *8B = 1KB row)
  g1[2] = (int)((unsigned)KT << 16); // tensor_dim0[31:16]=0 | tensor_dim1[15:0]=KT
  g1[3] = (int)(128u << 16);       // tensor_dim1[31:16]=0 | tile_dim0=128
  g1[4] = KT;                      // tile_dim1=KT, tile_dim2=0
  g1[5] = 13824;                   // tensor_dim0_stride = 110592B / 8B units
  g1[6] = 0;
  g1[7] = 0;
  i32x4 gz4 = {};
  i32x8 gz8 = {};
  __builtin_amdgcn_tensor_load_to_lds(g0, g1, gz4, gz4, gz8, 0);
#else
  // fallback: wave-0 cooperative copy, b128 per lane
  int lane = threadIdx.x & 31;
  for (int idx = lane; idx < KT * 64; idx += 32) {      // 64 x 16B chunks per row
    int kt = idx >> 6, ch = idx & 63;
    v8bf t = *(const v8bf*)(gsrc + (size_t)kt * ROWSTRIDE_USH + ch * 8);
    *(v8bf*)(smem + ldsUshortOff + (size_t)idx * 8) = t;
  }
#endif
}

__device__ __forceinline__ void tdm_wait() {
#ifdef USE_TDM
  __builtin_amdgcn_s_wait_tensorcnt(0);
#endif
}

// ---------------- weight repack: f32 row-major -> bf16 WMMA-B fragment-major ----
// dst[((kt*NT_FULL + nt)*32 + lane)*16 + slot]
//   lane<16 : K = kt*32 + slot ; lane>=16: K = kt*32 + 16 + slot ; N = nt*16 + (lane&15)
__global__ void repack_w0(const float* __restrict__ W0, unsigned short* __restrict__ dst) {
  const size_t TOT = W0R_ELEMS;
  size_t stride = (size_t)gridDim.x * blockDim.x;
  for (size_t i = (size_t)blockIdx.x * blockDim.x + threadIdx.x; i < TOT; i += stride) {
    int slot = (int)(i & 15);
    int lane = (int)((i >> 4) & 31);
    int frag = (int)(i >> 9);
    int nt = frag % NT_FULL;
    int kt = frag / NT_FULL;
    int k = kt * 32 + ((lane < 16) ? slot : 16 + slot);
    int ncol = nt * 16 + (lane & 15);
    int sc; bool cok;
    if (nt < NT_HALF) { sc = ncol;                 cok = (ncol < NHID); }
    else              { sc = NHID + (ncol - NPAD); cok = ((ncol - NPAD) < NHID); }
    int sk; bool kok;
    if (k < KPAD1) { sk = k;                  kok = (k < NHID); }
    else           { sk = NHID + (k - KPAD1); kok = ((k - KPAD1) < NHID); }
    float v = (cok && kok) ? W0[(size_t)sk * (2 * NHID) + sc] : 0.0f;
    dst[i] = f2bf(v);
  }
}

__global__ void repack_ws(const float* __restrict__ Ws, unsigned short* __restrict__ dst) {
  const size_t TOT = WSR_ELEMS;
  size_t stride = (size_t)gridDim.x * blockDim.x;
  for (size_t i = (size_t)blockIdx.x * blockDim.x + threadIdx.x; i < TOT; i += stride) {
    int e = (int)(i / WSR_PER);
    size_t r = i - (size_t)e * WSR_PER;
    int slot = (int)(r & 15);
    int lane = (int)((r >> 4) & 31);
    int frag = (int)(r >> 9);
    int nt = frag % NT_FULL;
    int kt = frag / NT_FULL;
    int k = kt * 32 + ((lane < 16) ? slot : 16 + slot);
    int ncol = nt * 16 + (lane & 15);
    int sc; bool cok;
    if (nt < NT_HALF) { sc = ncol;                 cok = (ncol < NHID); }
    else              { sc = NHID + (ncol - NPAD); cok = ((ncol - NPAD) < NHID); }
    bool kok = (k < NHID);
    float v = (cok && kok) ? Ws[((size_t)e * NHID + k) * (2 * NHID) + sc] : 0.0f;
    dst[i] = f2bf(v);
  }
}

// ---------------- one matmul+gating stage (B from LDS) ----------------
__device__ __forceinline__ void stage_mm_lds(
    const unsigned short* __restrict__ A, int Kpad, int KT,
    unsigned cOff, unsigned hOff,          // ushort offsets into dynamic LDS
    const float* __restrict__ Sprev,
    float* __restrict__ Sout, unsigned short* __restrict__ SoutBF,
    float* __restrict__ Acc, int accMode, int actCode,
    int mt, int np, int lane)
{
  extern __shared__ unsigned short smem[];
  const unsigned short* arow =
      A + (size_t)(mt * 16 + (lane & 15)) * Kpad + ((lane >> 4) << 4);
  const unsigned laneOff = (unsigned)lane * 16u;
  v8f accC = {};
  v8f accH = {};
  for (int kt = 0; kt < KT; ++kt) {
    union { v16bf v; v8bf h[2]; } ua, ub, uh;
    const unsigned short* ap = arow + kt * 32;
    __builtin_prefetch(ap + 64, 0, 1);                 // next-next A k-tile
    ua.h[0] = *(const v8bf*)(ap);
    ua.h[1] = *(const v8bf*)(ap + 8);
    const unsigned short* bp = smem + cOff + (unsigned)kt * FRAG_USH + laneOff;
    ub.h[0] = *(const v8bf*)(bp);
    ub.h[1] = *(const v8bf*)(bp + 8);
    const unsigned short* hp = smem + hOff + (unsigned)kt * FRAG_USH + laneOff;
    uh.h[0] = *(const v8bf*)(hp);
    uh.h[1] = *(const v8bf*)(hp + 8);
    accC = __builtin_amdgcn_wmma_f32_16x16x32_bf16(false, ua.v, false, ub.v,
                                                   (short)0, accC, false, false);
    accH = __builtin_amdgcn_wmma_f32_16x16x32_bf16(false, ua.v, false, uh.v,
                                                   (short)0, accH, false, false);
  }
  // C/D layout: VGPR r, lane l -> row = mt*16 + (l<16 ? r : r+8), col = (l&15)
  int j = np * 16 + (lane & 15);
  if (j < NHID) {
    int mbase = mt * 16 + ((lane >> 4) << 3);
#pragma unroll
    for (int r = 0; r < 8; ++r) {
      int m = mbase + r;
      float sp = Sprev[(size_t)m * NHID + j];
      float s = sp + sigf(accC[r]) * (actf(actCode, accH[r]) - sp);
      Sout[(size_t)m * NHID + j] = s;
      SoutBF[(size_t)m * NPAD + j] = f2bf(s);
      if (accMode == 1)      Acc[(size_t)m * NHID + j] = s;
      else if (accMode == 2) Acc[(size_t)m * NHID + j] += s;
    }
  }
}

// ---------------- persistent recurrent kernel ----------------
__global__ void __launch_bounds__(NTHREADS, 1)
darts_rnn(const float* __restrict__ x, const float* __restrict__ h0,
          const unsigned short* __restrict__ W0r, const unsigned short* __restrict__ Wsr,
          unsigned short* __restrict__ A0, float* __restrict__ hprevF,
          unsigned short* __restrict__ stateBF, float* __restrict__ stateF,
          float* __restrict__ accum, float* __restrict__ out,
          unsigned* __restrict__ syncCnt, unsigned* __restrict__ syncGen)
{
  const int tid   = threadIdx.x;
  const int gtid  = blockIdx.x * NTHREADS + tid;
  const int gsize = NBLOCKS * NTHREADS;
  const int lane  = tid & 31;
  const int np    = blockIdx.x % NT_HALF;          // column pair owned by this WG
  const int mt    = (blockIdx.x / NT_HALF) * 8 + (tid >> 5); // row tile per wave
  unsigned mygen = 0;

  // init h_prev from input hidden state
  for (int i = gtid; i < BH; i += gsize) {
    float h = h0[i];
    hprevF[i] = h;
    int b = i / NHID, j = i - b * NHID;
    A0[(size_t)b * KPAD0 + NPAD + j] = f2bf(h);
  }
  grid_sync(syncCnt, syncGen, mygen);

  for (int t = 0; t < T_STEPS; ++t) {
    // issue stage-0 B slab TDM (c half -> [0..), h half -> [KT0*512..)),
    // overlapped with x_t conversion
    if (tid < 32) {
      tdm_load_slab(W0r + (size_t)np * FRAG_USH, 0, KT0);
      tdm_load_slab(W0r + (size_t)(np + NT_HALF) * FRAG_USH, KT0 * FRAG_USH, KT0);
    }
    const float* xt = x + (size_t)t * BH;
    for (int i = gtid; i < BH; i += gsize) {
      int b = i / NHID, j = i - b * NHID;
      A0[(size_t)b * KPAD0 + j] = f2bf(xt[i]);
    }
    grid_sync(syncCnt, syncGen, mygen);

    // stage 0: s0 = h + sig(c0)*(tanh(h0) - h)
    if (tid < 32) tdm_wait();
    __syncthreads();
    stage_mm_lds(A0, KPAD0, KT0, 0, KT0 * FRAG_USH, hprevF,
                 stateF, stateBF, nullptr, /*accMode=*/0, /*tanh*/0, mt, np, lane);
    grid_sync(syncCnt, syncGen, mygen);

    // 8 DARTS edges, LDS double-buffered B slabs
    if (tid < 32) {
      const unsigned short* we = Wsr;                      // edge 0 weights
      tdm_load_slab(we + (size_t)np * FRAG_USH, 0, KT1);
      tdm_load_slab(we + (size_t)(np + NT_HALF) * FRAG_USH, EDGE_SLAB_USH, KT1);
    }
#pragma unroll
    for (int e = 0; e < 8; ++e) {
      const int pred[8] = {0, 1, 1, 1, 2, 5, 3, 5};
      const int actc[8] = {0, 1, 1, 3, 0, 2, 0, 1}; // tanh,relu,relu,id,tanh,sig,tanh,relu
      const unsigned buf = (unsigned)(e & 1) * EDGE_BUF_USH;
      if (tid < 32) tdm_wait();
      __syncthreads();
      if (e < 7 && tid < 32) {            // prefetch next edge into other buffer
        const unsigned short* wn = Wsr + (size_t)(e + 1) * WSR_PER;
        const unsigned nbuf = (unsigned)((e + 1) & 1) * EDGE_BUF_USH;
        tdm_load_slab(wn + (size_t)np * FRAG_USH, nbuf, KT1);
        tdm_load_slab(wn + (size_t)(np + NT_HALF) * FRAG_USH, nbuf + EDGE_SLAB_USH, KT1);
      }
      const unsigned short* Ap = stateBF + (size_t)pred[e] * BATCH * NPAD;
      const float*          Sp = stateF  + (size_t)pred[e] * BH;
      stage_mm_lds(Ap, NPAD, KT1, buf, buf + EDGE_SLAB_USH, Sp,
                   stateF + (size_t)(e + 1) * BH,
                   stateBF + (size_t)(e + 1) * BATCH * NPAD,
                   accum, (e == 0) ? 1 : 2, actc[e], mt, np, lane);
      grid_sync(syncCnt, syncGen, mygen);
    }

    // head: h_t = mean(states 1..8)
    for (int i = gtid; i < BH; i += gsize) {
      float h = accum[i] * 0.125f;
      out[(size_t)t * BH + i] = h;
      if (t == T_STEPS - 1) out[(size_t)T_STEPS * BH + i] = h;
      hprevF[i] = h;
      int b = i / NHID, j = i - b * NHID;
      A0[(size_t)b * KPAD0 + NPAD + j] = f2bf(h);
    }
    grid_sync(syncCnt, syncGen, mygen);
  }
}

// ---------------- launch ----------------
extern "C" void kernel_launch(void* const* d_in, const int* in_sizes, int n_in,
                              void* d_out, int out_size, void* d_ws, size_t ws_size,
                              hipStream_t stream) {
  const float* x  = (const float*)d_in[0];   // [70,256,850]
  const float* h0 = (const float*)d_in[1];   // [1,256,850]
  const float* W0 = (const float*)d_in[2];   // [1700,1700]
  const float* Ws = (const float*)d_in[3];   // [8,850,1700]
  float* out = (float*)d_out;

  uint8_t* p = (uint8_t*)d_ws;
  auto carve = [&](size_t bytes) -> uint8_t* {
    uint8_t* q = p;
    p += (bytes + 255) & ~(size_t)255;
    return q;
  };
  unsigned short* W0r     = (unsigned short*)carve(W0R_ELEMS * 2);
  unsigned short* Wsr     = (unsigned short*)carve(WSR_ELEMS * 2);
  unsigned short* A0      = (unsigned short*)carve((size_t)BATCH * KPAD0 * 2);
  unsigned short* stateBF = (unsigned short*)carve((size_t)9 * BATCH * NPAD * 2);
  float*          hprevF  = (float*)carve((size_t)BH * 4);
  float*          stateF  = (float*)carve((size_t)9 * BH * 4);
  float*          accum   = (float*)carve((size_t)BH * 4);
  unsigned*       sync    = (unsigned*)carve(256);

  (void)hipMemsetAsync(sync, 0, 256, stream);
  repack_w0<<<512, 256, 0, stream>>>(W0, W0r);
  repack_ws<<<1024, 256, 0, stream>>>(Ws, Wsr);
  darts_rnn<<<NBLOCKS, NTHREADS, LDS_BYTES, stream>>>(x, h0, W0r, Wsr, A0, hprevF,
                                                      stateBF, stateF, accum, out,
                                                      sync, sync + 64);
}
